// MultiHeadAttention_58961311039829
// MI455X (gfx1250) — compile-verified, matrix operand order checked by reading, not run on hardware
//
#include <hip/hip_runtime.h>
#include <hip/hip_bf16.h>
#include <stdint.h>

// ---------------------------------------------------------------------------
// MultiHeadAttention for MI455X (gfx1250): bf16 WMMA, f32 accumulate,
// async global->LDS tile streaming (ASYNCcnt) with double buffering,
// DPP16 VALU row reductions for the online softmax.
//   y = softmax_causal( (xWq^T)(xWk^T)^T * H^-0.5 ) (xWv^T)  -> @Wp^T + bp
// B=4 T=2048 K=1024 H=16 S=64
// ---------------------------------------------------------------------------

typedef __attribute__((ext_vector_type(16))) __bf16 v16bf;
typedef __attribute__((ext_vector_type(8)))  float  v8f;
typedef __attribute__((ext_vector_type(4)))  __bf16 v4bf;

static constexpr int BATCH = 4;
static constexpr int T     = 2048;
static constexpr int KDIM  = 1024;
static constexpr int NHEAD = 16;
static constexpr int HS    = 64;
static constexpr int MROWS = BATCH * T;   // 8192

__device__ __forceinline__ v8f wmma_bf16(v16bf a, v16bf b, v8f c) {
  // (neg_a, A, neg_b, B, c_mod, C, reuse_a, reuse_b)
  return __builtin_amdgcn_wmma_f32_16x16x32_bf16(false, a, false, b, (short)0, c,
                                                 false, false);
}

// Async global->LDS 16B copy per lane. LDS generic pointers carry the LDS
// byte address in their low 32 bits,
// which is exactly the VDST operand the async op wants.
__device__ __forceinline__ void async_ld_b128(const void* gsrc, void* ldst) {
  asm volatile("global_load_async_to_lds_b128 %0, %1, off"
               :: "v"((uint32_t)(uintptr_t)ldst), "v"(gsrc)
               : "memory");
}
__device__ __forceinline__ void wait_async0() {
  asm volatile("s_wait_asynccnt 0x0" ::: "memory");
}

// DPP16 cross-lane permute (VALU pipe, no LDS traffic).
template <int CTRL>
__device__ __forceinline__ float dpp_permf(float v) {
  return __uint_as_float((unsigned)__builtin_amdgcn_update_dpp(
      (int)__float_as_uint(v), (int)__float_as_uint(v), CTRL, 0xF, 0xF, false));
}
// Full 16-lane (half-wave) allreduce: quad xor1, quad xor2, row_ror:8, row_ror:4
__device__ __forceinline__ float red16_max(float v) {
  v = fmaxf(v, dpp_permf<0xB1>(v));    // quad_perm [1,0,3,2]  (xor 1)
  v = fmaxf(v, dpp_permf<0x4E>(v));    // quad_perm [2,3,0,1]  (xor 2)
  v = fmaxf(v, dpp_permf<0x128>(v));   // row_ror:8            (xor 8 in 16-row)
  v = fmaxf(v, dpp_permf<0x124>(v));   // row_ror:4            (folds last quads)
  return v;
}
__device__ __forceinline__ float red16_sum(float v) {
  v += dpp_permf<0xB1>(v);
  v += dpp_permf<0x4E>(v);
  v += dpp_permf<0x128>(v);
  v += dpp_permf<0x124>(v);
  return v;
}

// ------------------------------ f32 -> bf16 --------------------------------
__global__ __launch_bounds__(256) void cvt_f32_bf16(const float* __restrict__ in,
                                                    __bf16* __restrict__ out, int n) {
  int i = (blockIdx.x * 256 + threadIdx.x) * 4;
  if (i + 3 < n) {
    float4 v = *(const float4*)(in + i);
    v4bf o;
    o[0] = (__bf16)v.x; o[1] = (__bf16)v.y; o[2] = (__bf16)v.z; o[3] = (__bf16)v.w;
    *(v4bf*)(out + i) = o;
  }
}

// ------------------------------ GEMM: C = A * W^T --------------------------
// A: bf16 [MROWS][KDIM] row-major.  Wt: bf16 [out][in] row-major (= torch W).
// LDS B tile in [n][k] layout == direct row-major tile of W.
// BK=64, double-buffered async streaming; 16 barriers per block.
// MODE 0: f32 out [MROWS][KDIM] + bias
// MODE 1: bf16 out scattered to [B,H,T,S]
// MODE 2: bf16 out scattered to [B,H,S,T]   (transposed V for PV operand)
template <int MODE>
__global__ __launch_bounds__(256) void gemm_xwt(const __bf16* __restrict__ A,
                                                const __bf16* __restrict__ Wt,
                                                const float* __restrict__ bias,
                                                void* __restrict__ Cout) {
  constexpr int LP = 72;                 // 64 + 8 pad; 36*m mod 64 distinct
  __shared__ __bf16 sA[2][128 * LP];     // 2 x 18 KB
  __shared__ __bf16 sB[2][128 * LP];     // 2 x 18 KB  (72 KB total)

  const int tid  = threadIdx.x;
  const int wave = tid >> 5;
  const int lane = tid & 31;
  const int half = lane >> 4;
  const int ln   = lane & 15;
  const int row0 = blockIdx.x * 128;
  const int col0 = blockIdx.y * 128;
  const int wm   = (wave & 1) * 64;      // waves 2 (M) x 4 (N)
  const int wn   = (wave >> 1) * 32;

  // per-thread slice of a 128x64 tile: 4 x 16B (rows stride 32)
  const int lr = tid >> 3;               // 0..31
  const int lc = (tid & 7) * 8;          // 0..56

  v8f acc[4][2];
#pragma unroll
  for (int mi = 0; mi < 4; ++mi)
#pragma unroll
    for (int ni = 0; ni < 2; ++ni) acc[mi][ni] = (v8f){};

  auto stream_tile = [&](int k0, int nb) {
#pragma unroll
    for (int i = 0; i < 4; ++i) {
      int r = lr + i * 32;
      async_ld_b128(&A [(size_t)(row0 + r) * KDIM + k0 + lc], &sA[nb][r * LP + lc]);
      async_ld_b128(&Wt[(size_t)(col0 + r) * KDIM + k0 + lc], &sB[nb][r * LP + lc]);
    }
  };

  stream_tile(0, 0);
  constexpr int NT = KDIM / 64;          // 16 k-tiles
  for (int kt = 0; kt < NT; ++kt) {
    wait_async0();                       // my slice of tile kt is in LDS
    __syncthreads();                     // everyone's slice is; prev buf free
    const int buf = kt & 1;
    if (kt + 1 < NT) stream_tile((kt + 1) * 64, buf ^ 1);

#pragma unroll
    for (int kk = 0; kk < 2; ++kk) {     // two 32-wide WMMA k-substeps
      const int ko = kk * 32;
      v16bf af[4], bfr[2];
#pragma unroll
      for (int mi = 0; mi < 4; ++mi) {   // lanes0-15 K 0..7 & 16..23; +8 hi half
        int m = wm + mi * 16 + ln;
        uint4* p = (uint4*)&af[mi];
        p[0] = *(const uint4*)&sA[buf][m * LP + ko + half * 8];
        p[1] = *(const uint4*)&sA[buf][m * LP + ko + 16 + half * 8];
      }
#pragma unroll
      for (int ni = 0; ni < 2; ++ni) {   // B in [n][k] LDS layout, mirror gather
        int nn = wn + ni * 16 + ln;
        uint4* p = (uint4*)&bfr[ni];
        p[0] = *(const uint4*)&sB[buf][nn * LP + ko + half * 8];
        p[1] = *(const uint4*)&sB[buf][nn * LP + ko + 16 + half * 8];
      }
#pragma unroll
      for (int mi = 0; mi < 4; ++mi)
#pragma unroll
        for (int ni = 0; ni < 2; ++ni)
          acc[mi][ni] = wmma_bf16(af[mi], bfr[ni], acc[mi][ni]);
    }
  }

  // C layout: vgpr r holds row r (lanes0-15) / r+8 (lanes16-31), col = ln
#pragma unroll
  for (int mi = 0; mi < 4; ++mi)
#pragma unroll
    for (int ni = 0; ni < 2; ++ni)
#pragma unroll
      for (int r = 0; r < 8; ++r) {
        int gm = row0 + wm + mi * 16 + r + half * 8;
        int gn = col0 + wn + ni * 16 + ln;
        float v = acc[mi][ni][r];
        if constexpr (MODE == 0) {
          ((float*)Cout)[(size_t)gm * KDIM + gn] = v + bias[gn];
        } else {
          int b = gm >> 11, t = gm & (T - 1);
          int h = gn >> 6,  s = gn & (HS - 1);
          if constexpr (MODE == 1)
            ((__bf16*)Cout)[(((size_t)(b * NHEAD + h) * T + t) << 6) + s] = (__bf16)v;
          else
            ((__bf16*)Cout)[((size_t)(b * NHEAD + h) * HS + s) * T + t] = (__bf16)v;
        }
      }
}

// ------------------------------ attention ----------------------------------
// grid (T/128, B*H). Each wave: 16 query rows, full S=64 accumulator, online
// softmax over 64-wide key blocks up to the causal diagonal. K/V tiles are
// double-buffered and streamed with async loads; reductions run on DPP16.
__global__ __launch_bounds__(256) void attn_kernel(const __bf16* __restrict__ Qh,
                                                   const __bf16* __restrict__ Kh,
                                                   const __bf16* __restrict__ Vt,
                                                   __bf16* __restrict__ O) {
  constexpr int LP = 72;                 // 64 + 8 pad
  __shared__ __bf16 sK[2][64 * LP];      // [u][s]  -> B operand for Q*K^T
  __shared__ __bf16 sV[2][64 * LP];      // [s][u]  -> B operand for P*V
  __shared__ __bf16 sP[128 * LP];        // per-wave P tiles [m][u]

  const int tid  = threadIdx.x;
  const int wave = tid >> 5;
  const int lane = tid & 31;
  const int half = lane >> 4;
  const int ln   = lane & 15;
  const int t0   = blockIdx.x * 128;
  const int bh   = blockIdx.y;           // b*NHEAD + h
  const size_t base = (size_t)bh * T * HS;

  // per-thread slice of the 64x64 K/V tiles: 2 x 16B each
  const int kr0 = tid >> 3, kc0 = (tid & 7) * 8;          // idx = tid
  const int kr1 = (tid + 256) >> 3, kc1 = kc0;            // idx = tid + 256

  // Q: 16x64 per wave, kept in registers (2 A-fragments), loaded from global
  v16bf qf[2];
  const int tRow = t0 + wave * 16 + ln;  // A layout row m = ln
#pragma unroll
  for (int f = 0; f < 2; ++f) {
    uint4* p = (uint4*)&qf[f];
    p[0] = *(const uint4*)&Qh[base + (size_t)tRow * HS + f * 32 + half * 8];
    p[1] = *(const uint4*)&Qh[base + (size_t)tRow * HS + f * 32 + 16 + half * 8];
  }

  v8f acc[4];
#pragma unroll
  for (int st = 0; st < 4; ++st) acc[st] = (v8f){};
  float m_run[8], l_run[8];
#pragma unroll
  for (int r = 0; r < 8; ++r) { m_run[r] = -3.0e38f; l_run[r] = 0.f; }

  // prologue: stream key-block 0 into buffer 0
  async_ld_b128(&Kh[base + (size_t)kr0 * HS + kc0],       &sK[0][kr0 * LP + kc0]);
  async_ld_b128(&Vt[base + (size_t)kr0 * T + kc0],        &sV[0][kr0 * LP + kc0]);
  async_ld_b128(&Kh[base + (size_t)kr1 * HS + kc1],       &sK[0][kr1 * LP + kc1]);
  async_ld_b128(&Vt[base + (size_t)kr1 * T + kc1],        &sV[0][kr1 * LP + kc1]);

  const int njt = (t0 + 128) / 64;       // causal: key blocks up to block end
  for (int jt = 0; jt < njt; ++jt) {
    const int j0 = jt * 64;
    wait_async0();
    __syncthreads();
    const int buf = jt & 1;
    if (jt + 1 < njt) {                  // stream next K/V block
      const int jn = j0 + 64, nb = buf ^ 1;
      async_ld_b128(&Kh[base + (size_t)(jn + kr0) * HS + kc0], &sK[nb][kr0 * LP + kc0]);
      async_ld_b128(&Vt[base + (size_t)kr0 * T + jn + kc0],    &sV[nb][kr0 * LP + kc0]);
      async_ld_b128(&Kh[base + (size_t)(jn + kr1) * HS + kc1], &sK[nb][kr1 * LP + kc1]);
      async_ld_b128(&Vt[base + (size_t)kr1 * T + jn + kc1],    &sV[nb][kr1 * LP + kc1]);
    }

    // ---- scores: 4 u-subtiles, k = s (two 32-wide WMMA steps) ----
    v8f sc[4];
#pragma unroll
    for (int ut = 0; ut < 4; ++ut) {
      int n = ut * 16 + ln;
      v16bf b0, b1;
      uint4* p0 = (uint4*)&b0;
      p0[0] = *(const uint4*)&sK[buf][n * LP + half * 8];
      p0[1] = *(const uint4*)&sK[buf][n * LP + 16 + half * 8];
      uint4* p1 = (uint4*)&b1;
      p1[0] = *(const uint4*)&sK[buf][n * LP + 32 + half * 8];
      p1[1] = *(const uint4*)&sK[buf][n * LP + 48 + half * 8];
      v8f c = (v8f){};
      c = wmma_bf16(qf[0], b0, c);
      c = wmma_bf16(qf[1], b1, c);
      sc[ut] = c;
    }

    // ---- scale + causal mask + row max ----
    float bmax[8];
#pragma unroll
    for (int r = 0; r < 8; ++r) bmax[r] = -3.0e38f;
#pragma unroll
    for (int ut = 0; ut < 4; ++ut)
#pragma unroll
      for (int r = 0; r < 8; ++r) {
        float v = sc[ut][r] * 0.25f;     // H^-0.5 = 16^-0.5
        int trw = t0 + wave * 16 + r + half * 8;
        int u   = j0 + ut * 16 + ln;
        v = (u <= trw) ? v : -3.0e38f;
        sc[ut][r] = v;
        bmax[r] = fmaxf(bmax[r], v);
      }
#pragma unroll
    for (int r = 0; r < 8; ++r) bmax[r] = red16_max(bmax[r]);

    float alpha[8], rsum[8];
#pragma unroll
    for (int r = 0; r < 8; ++r) {
      float mnew = fmaxf(m_run[r], bmax[r]);
      alpha[r] = __expf(m_run[r] - mnew);
      m_run[r] = mnew;
      rsum[r] = 0.f;
    }
#pragma unroll
    for (int ut = 0; ut < 4; ++ut)
#pragma unroll
      for (int r = 0; r < 8; ++r) {
        float p = __expf(sc[ut][r] - m_run[r]);
        sc[ut][r] = p;
        rsum[r] += p;
      }
#pragma unroll
    for (int r = 0; r < 8; ++r)
      l_run[r] = l_run[r] * alpha[r] + red16_sum(rsum[r]);
#pragma unroll
    for (int st = 0; st < 4; ++st)
#pragma unroll
      for (int r = 0; r < 8; ++r) acc[st][r] *= alpha[r];

    // ---- relayout P (C-layout -> A-layout) via per-wave LDS bounce ----
#pragma unroll
    for (int ut = 0; ut < 4; ++ut)
#pragma unroll
      for (int r = 0; r < 8; ++r) {
        int m = wave * 16 + r + half * 8;
        sP[m * LP + ut * 16 + ln] = (__bf16)sc[ut][r];
      }
    asm volatile("s_wait_dscnt 0x0" ::: "memory");  // same-wave store->load

    v16bf pf[2];
#pragma unroll
    for (int f = 0; f < 2; ++f) {
      int m = wave * 16 + ln;
      uint4* p = (uint4*)&pf[f];
      p[0] = *(const uint4*)&sP[m * LP + f * 32 + half * 8];
      p[1] = *(const uint4*)&sP[m * LP + f * 32 + 16 + half * 8];
    }

    // ---- PV: 4 s-subtiles, k = u (two 32-wide WMMA steps) ----
#pragma unroll
    for (int st = 0; st < 4; ++st) {
      int srow = st * 16 + ln;
      v16bf b0, b1;
      uint4* p0 = (uint4*)&b0;
      p0[0] = *(const uint4*)&sV[buf][srow * LP + half * 8];
      p0[1] = *(const uint4*)&sV[buf][srow * LP + 16 + half * 8];
      uint4* p1 = (uint4*)&b1;
      p1[0] = *(const uint4*)&sV[buf][srow * LP + 32 + half * 8];
      p1[1] = *(const uint4*)&sV[buf][srow * LP + 48 + half * 8];
      acc[st] = wmma_bf16(pf[0], b0, acc[st]);
      acc[st] = wmma_bf16(pf[1], b1, acc[st]);
    }
  }

  // ---- normalize + write O as bf16 [B,T,K] row-major ----
  const int b = bh >> 4, h = bh & 15;
#pragma unroll
  for (int st = 0; st < 4; ++st)
#pragma unroll
    for (int r = 0; r < 8; ++r) {
      int t = t0 + wave * 16 + r + half * 8;
      float v = acc[st][r] / l_run[r];
      O[((size_t)(b * T + t)) * KDIM + h * HS + st * 16 + ln] = (__bf16)v;
    }
}

// ------------------------------ launcher -----------------------------------
extern "C" void kernel_launch(void* const* d_in, const int* in_sizes, int n_in,
                              void* d_out, int out_size, void* d_ws, size_t ws_size,
                              hipStream_t stream) {
  const float* x  = (const float*)d_in[0];
  const float* Wq = (const float*)d_in[1];
  const float* Wk = (const float*)d_in[2];
  const float* Wv = (const float*)d_in[3];
  const float* Wp = (const float*)d_in[4];
  const float* bp = (const float*)d_in[5];

  char* ws = (char*)d_ws;
  const size_t XB  = 0;                              // 16 MiB (also attn output O)
  const size_t WQB = XB  + (size_t)MROWS * KDIM * 2; // 4 x 2 MiB weights
  const size_t WKB = WQB + (size_t)KDIM * KDIM * 2;
  const size_t WVB = WKB + (size_t)KDIM * KDIM * 2;
  const size_t WPB = WVB + (size_t)KDIM * KDIM * 2;
  const size_t QH  = WPB + (size_t)KDIM * KDIM * 2;  // [B,H,T,S] 16 MiB
  const size_t KH  = QH  + (size_t)MROWS * KDIM * 2;
  const size_t VT  = KH  + (size_t)MROWS * KDIM * 2; // [B,H,S,T] 16 MiB
  // total ~72 MiB

  __bf16* xb  = (__bf16*)(ws + XB);
  __bf16* wqb = (__bf16*)(ws + WQB);
  __bf16* wkb = (__bf16*)(ws + WKB);
  __bf16* wvb = (__bf16*)(ws + WVB);
  __bf16* wpb = (__bf16*)(ws + WPB);
  __bf16* qh  = (__bf16*)(ws + QH);
  __bf16* kh  = (__bf16*)(ws + KH);
  __bf16* vt  = (__bf16*)(ws + VT);
  __bf16* ob  = (__bf16*)(ws + XB);                  // alias: xb dead after QKV

  const int nX = MROWS * KDIM;                       // 8388608
  const int nW = KDIM * KDIM;                        // 1048576
  cvt_f32_bf16<<<nX / 1024, 256, 0, stream>>>(x,  xb,  nX);
  cvt_f32_bf16<<<nW / 1024, 256, 0, stream>>>(Wq, wqb, nW);
  cvt_f32_bf16<<<nW / 1024, 256, 0, stream>>>(Wk, wkb, nW);
  cvt_f32_bf16<<<nW / 1024, 256, 0, stream>>>(Wv, wvb, nW);
  cvt_f32_bf16<<<nW / 1024, 256, 0, stream>>>(Wp, wpb, nW);

  dim3 gg(MROWS / 128, KDIM / 128);                  // 64 x 8
  gemm_xwt<1><<<gg, 256, 0, stream>>>(xb, wqb, nullptr, qh);
  gemm_xwt<1><<<gg, 256, 0, stream>>>(xb, wkb, nullptr, kh);
  gemm_xwt<2><<<gg, 256, 0, stream>>>(xb, wvb, nullptr, vt);

  attn_kernel<<<dim3(T / 128, BATCH * NHEAD), 256, 0, stream>>>(qh, kh, vt, ob);

  gemm_xwt<0><<<gg, 256, 0, stream>>>(ob, wpb, bp, (float*)d_out);
}
